// EncoderLayer_66297115181649
// MI455X (gfx1250) — compile-verified, matrix operand order checked by reading, not run on hardware
//
#include <hip/hip_runtime.h>

// Problem constants (B=4, S=2048): T = B*S = 8192 tokens.
#define DM   1024
#define DFF  4096
#define NH   16
#define DK   64
#define TOK  8192
#define SEQ  2048

typedef __attribute__((ext_vector_type(16))) __bf16 v16bf;
typedef __attribute__((ext_vector_type(8)))  __bf16 v8bf;
typedef __attribute__((ext_vector_type(8)))  float  v8f;
typedef __attribute__((ext_vector_type(4)))  unsigned int u32x4;
typedef __attribute__((ext_vector_type(8)))  int i32x8;
typedef __attribute__((ext_vector_type(4)))  int i32x4;

__device__ __forceinline__ unsigned short f2bf_bits(float f) {
  unsigned u = __float_as_uint(f);
  return (unsigned short)((u + 0x7FFFu + ((u >> 16) & 1u)) >> 16);  // RNE
}

__device__ __forceinline__ v8f v8f_zero() {
  v8f z;
#pragma unroll
  for (int i = 0; i < 8; ++i) z[i] = 0.0f;
  return z;
}

// 16-element bf16 fragment from two 16B chunks, matching the CDNA5 16-bit
// 16x32 layout: lane m=lane%16, g=lane/16; elems 0..7 -> K=8g.., 8..15 -> K=16+8g..
__device__ __forceinline__ v16bf load_frag(const unsigned short* p0,
                                           const unsigned short* p1) {
  v8bf lo = *(const v8bf*)p0;
  v8bf hi = *(const v8bf*)p1;
  v16bf r;
#pragma unroll
  for (int i = 0; i < 8; ++i) { r[i] = lo[i]; r[i + 8] = hi[i]; }
  return r;
}

__device__ __forceinline__ v8f wmma_bf16(v16bf a, v16bf b, v8f c) {
  return __builtin_amdgcn_wmma_f32_16x16x32_bf16(false, a, false, b,
                                                 (short)0, c, false, false);
}

// ---------------------------------------------------------------------------
// TDM: async 2D bf16 tile load (tile_k x tile_rows, row stride = stride_elems)
// from global to LDS. Descriptor per CDNA5 ISA D# groups 0/1 (2D, no padding,
// no multicast). 6-arg builtin (clang-23/therock-10 lane); TENSORcnt-tracked.
// ---------------------------------------------------------------------------
__device__ __forceinline__ void tdm_load_2d_bf16(unsigned lds_off,
                                                 const unsigned short* gptr,
                                                 unsigned tile_k,
                                                 unsigned tile_rows,
                                                 unsigned stride_elems) {
  unsigned long long ga = (unsigned long long)(const void*)gptr;
  u32x4 g0;
  g0[0] = 1u;                                   // count=1, user descriptor
  g0[1] = lds_off;                              // lds_addr (bytes)
  g0[2] = (unsigned)ga;                         // global_addr[31:0]
  g0[3] = (unsigned)((ga >> 32) & 0x01FFFFFFull) | 0x80000000u; // [56:32]|type=2
  i32x8 g1;
  g1[0] = (1 << 16);                            // wg_mask=0, data_size=1 (2B)
  g1[1] = (int)((tile_k & 0xFFFFu) << 16);      // tensor_dim0[15:0]
  g1[2] = (int)((tile_rows & 0xFFFFu) << 16);   // td0 hi=0 | tensor_dim1[15:0]
  g1[3] = (int)((tile_k & 0xFFFFu) << 16);      // td1 hi=0 | tile_dim0
  g1[4] = (int)(tile_rows & 0xFFFFu);           // tile_dim1 | tile_dim2=0
  g1[5] = (int)stride_elems;                    // tensor_dim0_stride[31:0]
  g1[6] = 0;                                    // stride hi | dim1_stride lo
  g1[7] = 0;
  i32x4 z4 = {0, 0, 0, 0};
  i32x8 z8;
#pragma unroll
  for (int i = 0; i < 8; ++i) z8[i] = 0;
  __builtin_amdgcn_tensor_load_to_lds(g0, g1, z4, z4, z8, 0);
}

__device__ __forceinline__ unsigned lds_offset_of(const void* p) {
  return (unsigned)(unsigned long long)p;  // low 32 bits = LDS byte offset
}

// ---------------------------------------------------------------------------
// f32 -> bf16 conversion
// ---------------------------------------------------------------------------
__global__ __launch_bounds__(256) void f32_to_bf16_kernel(
    const float* __restrict__ src, unsigned short* __restrict__ dst, int n) {
  int i = blockIdx.x * blockDim.x + threadIdx.x;
  int stride = gridDim.x * blockDim.x;
  for (; i < n; i += stride) dst[i] = f2bf_bits(src[i]);
}

// ---------------------------------------------------------------------------
// W [K][N] f32  ->  Wt [N][K] bf16   (LDS tile transpose, coalesced both sides)
// ---------------------------------------------------------------------------
__global__ __launch_bounds__(256) void transpose_w_bf16(
    const float* __restrict__ W, unsigned short* __restrict__ Wt,
    int K, int N) {
  __shared__ float t[32][33];
  const int k0 = blockIdx.y * 32, n0 = blockIdx.x * 32;
  const int r = threadIdx.x >> 5, c = threadIdx.x & 31;
#pragma unroll
  for (int i = 0; i < 4; ++i)
    t[r + i * 8][c] = W[(size_t)(k0 + r + i * 8) * N + n0 + c];
  __syncthreads();
#pragma unroll
  for (int i = 0; i < 4; ++i)
    Wt[(size_t)(n0 + r + i * 8) * K + k0 + c] = f2bf_bits(t[c][r + i * 8]);
}

// ---------------------------------------------------------------------------
// WMMA GEMM: C[M,N] = A(bf16 [M][K]) @ Wt(bf16 [N][K])^T, optional ReLU.
// Block tile 128x128, 8 waves of 32x64 (2x4 WMMA accum), K-step 32.
// Tiles staged by the Tensor Data Mover, double buffered: the hot loop is
// only ds_load_b128 fragment reads + v_wmma, with TDM prefetching overlap.
// ---------------------------------------------------------------------------
__global__ __launch_bounds__(256) void gemm_bf16_wmma(
    const unsigned short* __restrict__ A, const unsigned short* __restrict__ Wt,
    float* __restrict__ Cf, unsigned short* __restrict__ Cb,
    int M, int N, int K, int relu) {
  __shared__ alignas(16) unsigned short Asl[2][128 * 32];  // [m][k]
  __shared__ alignas(16) unsigned short Bsl[2][128 * 32];  // [n][k]

  const int tid   = threadIdx.x;
  const int lane  = tid & 31;
  const int wid   = tid >> 5;
  const int waveM = wid & 3;   // 4 groups of 32 rows
  const int waveN = wid >> 2;  // 2 groups of 64 cols
  const int g     = lane >> 4;
  const int ln16  = lane & 15;

  const int m0 = blockIdx.y * 128;
  const int n0 = blockIdx.x * 128;

  v8f acc[2][4];
#pragma unroll
  for (int i = 0; i < 2; ++i)
#pragma unroll
    for (int j = 0; j < 4; ++j) acc[i][j] = v8f_zero();

  // Prologue: wave 0 DMA-loads the first A/B tiles into buffer 0.
  if (wid == 0) {
    tdm_load_2d_bf16(lds_offset_of(&Asl[0][0]), &A[(size_t)m0 * K], 32, 128, K);
    tdm_load_2d_bf16(lds_offset_of(&Bsl[0][0]), &Wt[(size_t)n0 * K], 32, 128, K);
    __builtin_amdgcn_s_wait_tensorcnt(0);
  }
  __syncthreads();

  int p = 0;
  for (int kk = 0; kk < K; kk += 32) {
    // Async prefetch of the next K-slab into the other buffer (overlaps WMMA).
    if (kk + 32 < K && wid == 0) {
      tdm_load_2d_bf16(lds_offset_of(&Asl[p ^ 1][0]),
                       &A[(size_t)m0 * K + kk + 32], 32, 128, K);
      tdm_load_2d_bf16(lds_offset_of(&Bsl[p ^ 1][0]),
                       &Wt[(size_t)n0 * K + kk + 32], 32, 128, K);
    }

    v16bf af[2], bf[4];
#pragma unroll
    for (int mi = 0; mi < 2; ++mi) {
      int mrow = waveM * 32 + mi * 16 + ln16;
      af[mi] = load_frag(&Asl[p][mrow * 32 + 8 * g],
                         &Asl[p][mrow * 32 + 16 + 8 * g]);
    }
#pragma unroll
    for (int ni = 0; ni < 4; ++ni) {
      int nrow = waveN * 64 + ni * 16 + ln16;
      bf[ni] = load_frag(&Bsl[p][nrow * 32 + 8 * g],
                         &Bsl[p][nrow * 32 + 16 + 8 * g]);
    }
#pragma unroll
    for (int mi = 0; mi < 2; ++mi)
#pragma unroll
      for (int ni = 0; ni < 4; ++ni)
        acc[mi][ni] = wmma_bf16(af[mi], bf[ni], acc[mi][ni]);

    if (wid == 0) __builtin_amdgcn_s_wait_tensorcnt(0);
    __syncthreads();
    p ^= 1;
  }

  // Epilogue: C layout = (row = r + 8*(lane/16), col = lane%16).
#pragma unroll
  for (int mi = 0; mi < 2; ++mi)
#pragma unroll
    for (int ni = 0; ni < 4; ++ni)
#pragma unroll
      for (int r = 0; r < 8; ++r) {
        int grow = m0 + waveM * 32 + mi * 16 + r + 8 * g;
        int gcol = n0 + waveN * 64 + ni * 16 + ln16;
        float v = acc[mi][ni][r];
        if (relu) v = fmaxf(v, 0.0f);
        if (Cf) Cf[(size_t)grow * N + gcol] = v;
        if (Cb) Cb[(size_t)grow * N + gcol] = f2bf_bits(v);
      }
}

// ---------------------------------------------------------------------------
// Flash attention: one block = (128 q-rows, one head). 8 waves x 16 q-rows.
// Streams K/V in 32-token tiles; online softmax; WMMA for QK^T and PV.
// ---------------------------------------------------------------------------
__global__ __launch_bounds__(256) void flash_attn_wmma(
    const unsigned short* __restrict__ Q, const unsigned short* __restrict__ Kd,
    const unsigned short* __restrict__ V, unsigned short* __restrict__ O) {
  __shared__ alignas(16) unsigned short Ksl[32 * 64];    // [kv][d]
  __shared__ alignas(16) unsigned short Vsl[64 * 32];    // [d][kv]  (V^T)
  __shared__ alignas(16) unsigned short Psl[8][16 * 32]; // per-wave P scratch

  const int tid  = threadIdx.x;
  const int lane = tid & 31;
  const int wid  = tid >> 5;
  const int g    = lane >> 4;
  const int ln16 = lane & 15;

  const int q0     = blockIdx.x * 128;
  const int hoff   = blockIdx.y * DK;
  const int batch0 = (q0 / SEQ) * SEQ;

  // Q fragments (16 rows x 64 d = two 16x32 A fragments) kept in registers.
  const int qrow = q0 + wid * 16 + ln16;
  v16bf qf[2];
#pragma unroll
  for (int c = 0; c < 2; ++c) {
    const unsigned short* qp = &Q[(size_t)qrow * DM + hoff + c * 32];
    qf[c] = load_frag(qp + 8 * g, qp + 16 + 8 * g);
  }

  v8f o[4];
#pragma unroll
  for (int t = 0; t < 4; ++t) o[t] = v8f_zero();
  float rm[8], rs[8];
#pragma unroll
  for (int r = 0; r < 8; ++r) { rm[r] = -1e30f; rs[r] = 0.0f; }

  const int skv = tid >> 3, sd = (tid & 7) * 8;   // K staging map
  const int tkv = tid & 31, td = (tid >> 5) * 8;  // V staging map

  for (int j = 0; j < SEQ; j += 32) {
    // Stage K tile [32 kv][64 d]: one 16B vector copy per thread.
    *(uint4*)&Ksl[skv * 64 + sd] =
        *(const uint4*)&Kd[(size_t)(batch0 + j + skv) * DM + hoff + sd];
    // Stage V transposed: one b128 global load, 8 scalar LDS scatter stores.
    uint4 vv = *(const uint4*)&V[(size_t)(batch0 + j + tkv) * DM + hoff + td];
    const unsigned short* pv = (const unsigned short*)&vv;
#pragma unroll
    for (int i = 0; i < 8; ++i) Vsl[(td + i) * 32 + tkv] = pv[i];
    __syncthreads();

    // S(16x32) = Q(16x64) @ K^T(64x32): 2 N-tiles x 2 K-chunks = 4 WMMAs.
    v8f sacc[2];
#pragma unroll
    for (int nt = 0; nt < 2; ++nt) {
      sacc[nt] = v8f_zero();
      int kvrow = nt * 16 + ln16;
#pragma unroll
      for (int c = 0; c < 2; ++c) {
        v16bf bfK = load_frag(&Ksl[kvrow * 64 + c * 32 + 8 * g],
                              &Ksl[kvrow * 64 + c * 32 + 16 + 8 * g]);
        sacc[nt] = wmma_bf16(qf[c], bfK, sacc[nt]);
      }
    }

    // Online softmax per row (rows live in 16-lane halves -> xor 1..8).
#pragma unroll
    for (int r = 0; r < 8; ++r) {
      float s0 = sacc[0][r] * 0.125f;  // 1/sqrt(DK)
      float s1 = sacc[1][r] * 0.125f;
      float m = fmaxf(s0, s1);
#pragma unroll
      for (int msk = 1; msk < 16; msk <<= 1)
        m = fmaxf(m, __shfl_xor(m, msk, 32));
      float mnew = fmaxf(rm[r], m);
      float corr = __expf(rm[r] - mnew);
      rm[r] = mnew;
      float p0 = __expf(s0 - mnew);
      float p1 = __expf(s1 - mnew);
      float rsum = p0 + p1;
#pragma unroll
      for (int msk = 1; msk < 16; msk <<= 1)
        rsum += __shfl_xor(rsum, msk, 32);
      rs[r] = rs[r] * corr + rsum;
#pragma unroll
      for (int t = 0; t < 4; ++t) o[t][r] *= corr;
      int prow = r + 8 * g;  // C-layout -> A-layout staging via LDS
      Psl[wid][prow * 32 + ln16]      = f2bf_bits(p0);
      Psl[wid][prow * 32 + 16 + ln16] = f2bf_bits(p1);
    }

    asm volatile("s_wait_dscnt 0" ::: "memory");  // wave-local P RAW fence

    // O(16x64) += P(16x32) @ V(32x64): 4 WMMAs.
    v16bf pf = load_frag(&Psl[wid][ln16 * 32 + 8 * g],
                         &Psl[wid][ln16 * 32 + 16 + 8 * g]);
#pragma unroll
    for (int dt = 0; dt < 4; ++dt) {
      int drow = dt * 16 + ln16;
      v16bf bfV = load_frag(&Vsl[drow * 32 + 8 * g],
                            &Vsl[drow * 32 + 16 + 8 * g]);
      o[dt] = wmma_bf16(pf, bfV, o[dt]);
    }
    __syncthreads();
  }

  // Normalize and store bf16 attention output.
#pragma unroll
  for (int r = 0; r < 8; ++r) {
    float inv = 1.0f / rs[r];
    int row = q0 + wid * 16 + r + 8 * g;
#pragma unroll
    for (int dt = 0; dt < 4; ++dt)
      O[(size_t)row * DM + hoff + dt * 16 + ln16] = f2bf_bits(o[dt][r] * inv);
  }
}

// ---------------------------------------------------------------------------
// y = LayerNorm(X + Delta) * gamma + beta ; optional f32 and bf16 outputs.
// ---------------------------------------------------------------------------
__global__ __launch_bounds__(256) void residual_layernorm(
    const float* __restrict__ X, const float* __restrict__ Dl,
    const float* __restrict__ gamma, const float* __restrict__ beta,
    float* __restrict__ Of, unsigned short* __restrict__ Ob) {
  __shared__ float red[256];
  const int row = blockIdx.x;
  const int tid = threadIdx.x;
  float v[4];
  float s = 0.0f;
#pragma unroll
  for (int i = 0; i < 4; ++i) {
    int idx = tid + i * 256;
    float t = X[(size_t)row * DM + idx] + Dl[(size_t)row * DM + idx];
    v[i] = t;
    s += t;
  }
  red[tid] = s;
  __syncthreads();
  for (int off = 128; off > 0; off >>= 1) {
    if (tid < off) red[tid] += red[tid + off];
    __syncthreads();
  }
  float mu = red[0] * (1.0f / DM);
  __syncthreads();
  float q = 0.0f;
#pragma unroll
  for (int i = 0; i < 4; ++i) { float d = v[i] - mu; q += d * d; }
  red[tid] = q;
  __syncthreads();
  for (int off = 128; off > 0; off >>= 1) {
    if (tid < off) red[tid] += red[tid + off];
    __syncthreads();
  }
  float rstd = rsqrtf(red[0] * (1.0f / DM) + 1e-5f);
#pragma unroll
  for (int i = 0; i < 4; ++i) {
    int idx = tid + i * 256;
    float y = (v[i] - mu) * rstd * gamma[idx] + beta[idx];
    if (Of) Of[(size_t)row * DM + idx] = y;
    if (Ob) Ob[(size_t)row * DM + idx] = f2bf_bits(y);
  }
}

// ---------------------------------------------------------------------------
extern "C" void kernel_launch(void* const* d_in, const int* in_sizes, int n_in,
                              void* d_out, int out_size, void* d_ws,
                              size_t ws_size, hipStream_t stream) {
  (void)in_sizes; (void)n_in; (void)out_size; (void)ws_size;

  const float* x    = (const float*)d_in[0];
  const float* Wq   = (const float*)d_in[1];
  const float* Wk   = (const float*)d_in[2];
  const float* Wv   = (const float*)d_in[3];
  const float* Wo   = (const float*)d_in[4];
  const float* Wf1  = (const float*)d_in[5];
  const float* Wf2  = (const float*)d_in[6];
  const float* ln1g = (const float*)d_in[7];
  const float* ln1b = (const float*)d_in[8];
  const float* ln2g = (const float*)d_in[9];
  const float* ln2b = (const float*)d_in[10];
  float* out = (float*)d_out;

  // Workspace layout (~248 MB).
  unsigned short* Xb   = (unsigned short*)d_ws;           // [TOK,DM] bf16
  unsigned short* Qb   = Xb  + (size_t)TOK * DM;
  unsigned short* Kb   = Qb  + (size_t)TOK * DM;
  unsigned short* Vb   = Kb  + (size_t)TOK * DM;
  unsigned short* Ab   = Vb  + (size_t)TOK * DM;          // attention out bf16
  unsigned short* Hb   = Ab  + (size_t)TOK * DM;          // h bf16
  unsigned short* FFb  = Hb  + (size_t)TOK * DM;          // [TOK,DFF] bf16
  float* Of = (float*)(FFb + (size_t)TOK * DFF);          // [TOK,DM] f32
  float* Hf = Of + (size_t)TOK * DM;                      // [TOK,DM] f32
  unsigned short* Wqt  = (unsigned short*)(Hf + (size_t)TOK * DM); // [DM][DM]
  unsigned short* Wkt  = Wqt + (size_t)DM * DM;
  unsigned short* Wvt  = Wkt + (size_t)DM * DM;
  unsigned short* Wot  = Wvt + (size_t)DM * DM;
  unsigned short* Wf1t = Wot + (size_t)DM * DM;           // [DFF][DM]
  unsigned short* Wf2t = Wf1t + (size_t)DM * DFF;         // [DM][DFF]

  // 0) one-time precision/layout conversions
  f32_to_bf16_kernel<<<2048, 256, 0, stream>>>(x, Xb, TOK * DM);
  dim3 gtq(DM / 32, DM / 32);
  transpose_w_bf16<<<gtq, 256, 0, stream>>>(Wq, Wqt, DM, DM);
  transpose_w_bf16<<<gtq, 256, 0, stream>>>(Wk, Wkt, DM, DM);
  transpose_w_bf16<<<gtq, 256, 0, stream>>>(Wv, Wvt, DM, DM);
  transpose_w_bf16<<<gtq, 256, 0, stream>>>(Wo, Wot, DM, DM);
  transpose_w_bf16<<<dim3(DFF / 32, DM / 32), 256, 0, stream>>>(Wf1, Wf1t, DM, DFF);
  transpose_w_bf16<<<dim3(DM / 32, DFF / 32), 256, 0, stream>>>(Wf2, Wf2t, DFF, DM);

  // 1) Q/K/V projections (bf16 outputs feed attention)
  dim3 gdd(DM / 128, TOK / 128);
  gemm_bf16_wmma<<<gdd, 256, 0, stream>>>(Xb, Wqt, nullptr, Qb, TOK, DM, DM, 0);
  gemm_bf16_wmma<<<gdd, 256, 0, stream>>>(Xb, Wkt, nullptr, Kb, TOK, DM, DM, 0);
  gemm_bf16_wmma<<<gdd, 256, 0, stream>>>(Xb, Wvt, nullptr, Vb, TOK, DM, DM, 0);

  // 2) flash attention (per-head)
  flash_attn_wmma<<<dim3(TOK / 128, NH), 256, 0, stream>>>(Qb, Kb, Vb, Ab);

  // 3) output projection, then h = LN(x + attn)
  gemm_bf16_wmma<<<gdd, 256, 0, stream>>>(Ab, Wot, Of, nullptr, TOK, DM, DM, 0);
  residual_layernorm<<<TOK, 256, 0, stream>>>(x, Of, ln1g, ln1b, Hf, Hb);

  // 4) FFN: relu(h @ Wf1) @ Wf2, then out = LN(h + ffn)
  gemm_bf16_wmma<<<dim3(DFF / 128, TOK / 128), 256, 0, stream>>>(
      Hb, Wf1t, nullptr, FFb, TOK, DFF, DM, 1);
  gemm_bf16_wmma<<<gdd, 256, 0, stream>>>(FFb, Wf2t, Of, nullptr, TOK, DM, DFF, 0);
  residual_layernorm<<<TOK, 256, 0, stream>>>(Hf, Of, ln2g, ln2b, out, nullptr);
}
